// GGMLLinear_72799695667549
// MI455X (gfx1250) — matrix operand order repaired and checked by reading, not compiled, not run on hardware
//
#include <hip/hip_runtime.h>

typedef __attribute__((ext_vector_type(16))) _Float16 v16h;
typedef __attribute__((ext_vector_type(8)))  float    v8f;
typedef __attribute__((ext_vector_type(4)))  int      i32x4;

#define IN_DIM   4096
#define OUT_DIM  14336
#define QBLOCK   32
#define NB       (IN_DIM / QBLOCK)     // 128 blocks per row
#define WAVES    4
#define KB_PER_WAVE (NB / WAVES)       // 32 blocks per wave
#define PF_DIST  4                     // prefetch distance in 32-K blocks (512B/row)

__global__ __launch_bounds__(WAVES * 32) void q8_gemv_wmma(
    const float* __restrict__ x,
    const int*   __restrict__ q,
    const float* __restrict__ scales,
    const float* __restrict__ bias,
    float*       __restrict__ out)
{
    __shared__ alignas(32) _Float16 xh[IN_DIM];     // f16 copy of activations
    __shared__ float part[WAVES][16];               // split-K partials

    const int tid = threadIdx.x;

    // Convert x (fp32) -> f16 into LDS once per workgroup (16 KB read; negligible
    // vs the 235 MB of q traffic, and it turns the B operand into two aligned
    // ds_read_b128 per WMMA).
    for (int i = tid; i < IN_DIM / 4; i += WAVES * 32) {
        const float4 v = ((const float4*)x)[i];
        xh[4 * i + 0] = (_Float16)v.x;
        xh[4 * i + 1] = (_Float16)v.y;
        xh[4 * i + 2] = (_Float16)v.z;
        xh[4 * i + 3] = (_Float16)v.w;
    }
    __syncthreads();

    const int wave = tid >> 5;
    const int lane = tid & 31;
    const int m    = lane & 15;        // A-matrix row within tile
    const int g    = lane >> 4;        // lane group (K-half select)
    const int tile = blockIdx.x;       // 16-row output tile
    const int row  = tile * 16 + m;

    const int*   qrow = q      + (size_t)row * IN_DIM;
    const float* srow = scales + (size_t)row * NB;

    v8f acc0 = {};
    v8f acc1 = {};

    const int kbase0 = wave * KB_PER_WAVE;

    for (int kk = 0; kk < KB_PER_WAVE; kk += 2) {
#pragma unroll
        for (int u = 0; u < 2; ++u) {
            const int b     = kbase0 + kk + u;
            const int kbase = b * QBLOCK;
            // A-matrix 16x32 layout (contiguous, matching B):
            //   lanes 0-15  (g=0): K=0..15;  lanes 16-31 (g=1): K=16..31
            const int off1 = kbase + 16 * g;       // 32B-aligned
            const int off2 = kbase + 16 * g + 8;   // 32B-aligned
            const i32x4 q0 = *(const i32x4*)(qrow + off1);
            const i32x4 q1 = *(const i32x4*)(qrow + off1 + 4);
            const i32x4 q2 = *(const i32x4*)(qrow + off2);
            const i32x4 q3 = *(const i32x4*)(qrow + off2 + 4);
            const float s  = srow[b];              // per-(row,block) Q8_0 scale

            // Stream-prefetch this row's q data a few blocks ahead
            // (single-pass 235 MB stream; helps hide HBM latency).
            if (kk + u + PF_DIST < KB_PER_WAVE)
                __builtin_prefetch(qrow + kbase + PF_DIST * QBLOCK + 8 * g, 0, 0);

            // Convert q -> f16 (exact: |q| <= 127), then fold the per-block
            // scale in with packed f16 multiplies (v_pk_mul_f16). The f32
            // accumulator then carries sum_b sum_i (s_b*q)*x directly.
            v16h a;
            a[0]  = (_Float16)(float)q0.x;
            a[1]  = (_Float16)(float)q0.y;
            a[2]  = (_Float16)(float)q0.z;
            a[3]  = (_Float16)(float)q0.w;
            a[4]  = (_Float16)(float)q1.x;
            a[5]  = (_Float16)(float)q1.y;
            a[6]  = (_Float16)(float)q1.z;
            a[7]  = (_Float16)(float)q1.w;
            a[8]  = (_Float16)(float)q2.x;
            a[9]  = (_Float16)(float)q2.y;
            a[10] = (_Float16)(float)q2.z;
            a[11] = (_Float16)(float)q2.w;
            a[12] = (_Float16)(float)q3.x;
            a[13] = (_Float16)(float)q3.y;
            a[14] = (_Float16)(float)q3.z;
            a[15] = (_Float16)(float)q3.w;

            const _Float16 hs = (_Float16)s;
            const v16h sv = {hs, hs, hs, hs, hs, hs, hs, hs,
                             hs, hs, hs, hs, hs, hs, hs, hs};
            a = a * sv;   // packed f16: 8x v_pk_mul_f16

            // B 32x16 in this code's contiguous k order: VGPR v, lanes 0-15 hold K=2v,2v+1; lanes 16-31
            // hold K=16+2v,16+2v+1. Batch=1 -> all columns identical, so the
            // per-lane 16-half operand is the contiguous run xh[kbase+16g ..].
            const v16h bm = *(const v16h*)&xh[kbase + 16 * g];

            if (u == 0)
                acc0 = __builtin_amdgcn_wmma_f32_16x16x32_f16(
                    false, a, false, bm, (short)0, acc0, false, false);
            else
                acc1 = __builtin_amdgcn_wmma_f32_16x16x32_f16(
                    false, a, false, bm, (short)0, acc1, false, false);
        }
    }

    const v8f acc = acc0 + acc1;

    // D layout: VGPR r -> (M=r, N=lane) for lanes 0-15, (M=8+r, N=lane-16) for
    // lanes 16-31. All N columns are identical; lanes with (lane&15)==0 hold
    // the N=0 column for rows r and 8+r respectively.
    if (m == 0) {
#pragma unroll
        for (int r = 0; r < 8; ++r)
            part[wave][r + 8 * g] = acc[r];
    }
    __syncthreads();

    if (tid < 16) {
        float sum = bias[tile * 16 + tid];
#pragma unroll
        for (int w = 0; w < WAVES; ++w)
            sum += part[w][tid];
        out[tile * 16 + tid] = sum;
    }
}

extern "C" void kernel_launch(void* const* d_in, const int* in_sizes, int n_in,
                              void* d_out, int out_size, void* d_ws, size_t ws_size,
                              hipStream_t stream) {
    const float* x      = (const float*)d_in[0];
    const int*   q      = (const int*)d_in[1];
    const float* scales = (const float*)d_in[2];
    const float* bias   = (const float*)d_in[3];
    float*       out    = (float*)d_out;

    dim3 grid(OUT_DIM / 16);
    dim3 block(WAVES * 32);
    hipLaunchKernelGGL(q8_gemv_wmma, grid, block, 0, stream, x, q, scales, bias, out);
}
